// KnowRGFD_54030688584334
// MI455X (gfx1250) — matrix operand order repaired and checked by reading, not compiled, not run on hardware
//
#include <hip/hip_runtime.h>
#include <math.h>

#define NN    50000
#define EE    800000
#define HH    128
#define RREL  5
#define CC    4
#define KTOT  ((RREL + 1) * HH)   // 768
#define NC    (NN * CC)

typedef __attribute__((ext_vector_type(16))) __bf16 v16bf;
typedef __attribute__((ext_vector_type(8)))  __bf16 v8bf;
typedef __attribute__((ext_vector_type(8)))  float  v8f;

static __device__ __forceinline__ unsigned short f2bf(float f) {
    unsigned u = __float_as_uint(f);
    unsigned r = ((u >> 16) & 1u) + 0x7FFFu;   // round-to-nearest-even
    return (unsigned short)((u + r) >> 16);
}

// ---------------- counts ----------------
__global__ __launch_bounds__(256) void count_k(const int* __restrict__ dst,
                                               const int* __restrict__ et,
                                               float* __restrict__ cnt) {
    int e = blockIdx.x * 256 + threadIdx.x;
    if (e < EE) atomicAdd(&cnt[dst[e] * RREL + et[e]], 1.0f);
}

__global__ __launch_bounds__(256) void inv_k(float* __restrict__ cnt) {
    int i = blockIdx.x * 256 + threadIdx.x;
    if (i < NN * RREL) {
        float c = cnt[i];
        cnt[i] = 1.0f / (c > 1.0f ? c : 1.0f);
    }
}

// ---------------- edge scatter: msg[dst*R+etype] += h[src] ----------------
__global__ __launch_bounds__(256) void scatter_k(const float* __restrict__ h,
                                                 const int* __restrict__ src,
                                                 const int* __restrict__ dst,
                                                 const int* __restrict__ et,
                                                 float* __restrict__ msg) {
    int e = blockIdx.x * 8 + (threadIdx.x >> 5);
    if (e >= EE) return;
    int lane = threadIdx.x & 31;
    int s   = src[e];
    int seg = dst[e] * RREL + et[e];
    const float4 v = *(const float4*)(h + (size_t)s * HH + lane * 4);
    float* mp = msg + (size_t)seg * HH + lane * 4;
    atomicAdd(mp + 0, v.x);
    atomicAdd(mp + 1, v.y);
    atomicAdd(mp + 2, v.z);
    atomicAdd(mp + 3, v.w);
}

// ---------------- pack W_cat [768x128] into per-lane WMMA B fragments ----------------
// flat index t = kchunk*4096 + ntile*512 + lane*16 + i   (bf16 bits in ushort)
// lane<16 : element i -> (K = kc*32 + i,      Ncol = ntile*16 + lane)
// lane>=16: element i -> (K = kc*32 + 16 + i, Ncol = ntile*16 + lane-16)
__global__ __launch_bounds__(256) void wfrag_k(const float* __restrict__ root,
                                               const float* __restrict__ W,
                                               unsigned short* __restrict__ Bfrag) {
    int t = blockIdx.x * 256 + threadIdx.x;
    if (t >= KTOT * HH) return;
    int i    = t & 15;
    int lane = (t >> 4) & 31;
    int w    = (t >> 9) & 7;
    int kc   = t >> 12;
    int K    = kc * 32 + ((lane >> 4) << 4) + i;
    int Ncol = w * 16 + (lane & 15);
    float val;
    if (K < HH) {
        val = root[K * HH + Ncol];
    } else {
        int rel = (K - HH) >> 7;
        int f   = (K - HH) & 127;
        val = W[((size_t)rel * HH + f) * HH + Ncol];
    }
    Bfrag[t] = f2bf(val);
}

// ---------------- fused RGCN layer GEMM: relu([h | msg*inv] @ Wcat + bias) ----------------
__global__ __launch_bounds__(256) void rgcn_gemm_k(const float* __restrict__ h,
                                                   const float* __restrict__ msg,
                                                   const float* __restrict__ invc,
                                                   const unsigned short* __restrict__ Bfrag,
                                                   const float* __restrict__ bias,
                                                   float* __restrict__ hout) {
    __shared__ __align__(16) unsigned short As[16 * 32];  // 16 rows x 32 k, bf16 bits
    const int tid  = threadIdx.x;
    const int wave = tid >> 5;
    const int lane = tid & 31;
    const int row0 = blockIdx.x * 16;   // N = 50000 = 3125 * 16 (exact)
    const int nb   = wave * 16;

    v8f acc = {};

    for (int kc = 0; kc < KTOT; kc += 32) {
        // stage A tile: 512 bf16, 2 per thread, fuse inv_cnt scaling
        {
            int idx = tid * 2;
            int r   = idx >> 5;
            int k   = idx & 31;
            int gk  = kc + k;
            int n   = row0 + r;
            float v0, v1;
            if (gk < HH) {
                const float* hp = h + (size_t)n * HH + gk;
                v0 = hp[0]; v1 = hp[1];
            } else {
                int rel = (gk - HH) >> 7;
                int f   = (gk - HH) & 127;
                float s = invc[n * RREL + rel];
                const float* mp = msg + ((size_t)n * RREL + rel) * HH + f;
                v0 = mp[0] * s; v1 = mp[1] * s;
            }
            As[r * 32 + k]     = f2bf(v0);
            As[r * 32 + k + 1] = f2bf(v1);
        }
        __syncthreads();

        // A fragment, contiguous K order matching Bfrag:
        //  lane<16 : K in {khi..khi+15}, khi=0 ; lane>=16: khi=16
        int arow = lane & 15;
        int khi  = (lane >> 4) << 4;
        v8bf alo = *(const v8bf*)&As[arow * 32 + khi];
        v8bf ahi = *(const v8bf*)&As[arow * 32 + khi + 8];
        union { v16bf v; v8bf h2[2]; } au;
        au.h2[0] = alo; au.h2[1] = ahi;

        // B fragment: one coalesced 32B load per lane
        size_t bidx = ((((size_t)(kc >> 5)) * 8 + wave) * 32 + lane) * 16;
        v16bf b = *(const v16bf*)(Bfrag + bidx);
        if (kc + 32 < KTOT)
            __builtin_prefetch(Bfrag + bidx + (size_t)8 * 32 * 16, 0, 1);

        acc = __builtin_amdgcn_wmma_f32_16x16x32_bf16(
            false, au.v, false, b, (short)0, acc, false, false);
        __syncthreads();
    }

    // C/D layout: VGPR v -> (M=v, N=lane) lanes 0-15 ; (M=v+8, N=lane-16) lanes 16-31
    int ncol  = nb + (lane & 15);
    int mbase = (lane >> 4) * 8;
    float bsv = bias[ncol];
#pragma unroll
    for (int v = 0; v < 8; ++v) {
        int m = row0 + mbase + v;
        float val = acc[v] + bsv;
        hout[(size_t)m * HH + ncol] = val > 0.0f ? val : 0.0f;
    }
}

// ---------------- layer attention: feat = sum_l softmax_l(h_last.aw + ab) * h_l ----------------
__global__ __launch_bounds__(256) void attn_k(const float* __restrict__ h1,
                                              const float* __restrict__ h2,
                                              const float* __restrict__ h3,
                                              const float* __restrict__ aw,
                                              const float* __restrict__ ab,
                                              float* __restrict__ feat, int L) {
    int n = blockIdx.x * 8 + (threadIdx.x >> 5);
    if (n >= NN) return;
    int lane = threadIdx.x & 31;
    const float* last = (L == 3) ? h3 : h2;
    float4 xl = *(const float4*)(last + (size_t)n * HH + lane * 4);

    float sc[3];
    float mx = -1e30f;
    for (int l = 0; l < L; ++l) {
        float4 w = *(const float4*)(aw + l * HH + lane * 4);
        float p = xl.x * w.x + xl.y * w.y + xl.z * w.z + xl.w * w.w;
        for (int o = 16; o > 0; o >>= 1) p += __shfl_xor(p, o, 32);
        p += ab[l];
        sc[l] = p;
        mx = p > mx ? p : mx;
    }
    float den = 0.0f;
    for (int l = 0; l < L; ++l) { sc[l] = __expf(sc[l] - mx); den += sc[l]; }
    float inv = 1.0f / den;

    const float* hs[3] = { h1, h2, h3 };
    float4 acc = { 0.f, 0.f, 0.f, 0.f };
    for (int l = 0; l < L; ++l) {
        float4 v = *(const float4*)(hs[l] + (size_t)n * HH + lane * 4);
        float s = sc[l] * inv;
        acc.x += s * v.x; acc.y += s * v.y; acc.z += s * v.z; acc.w += s * v.w;
    }
    *(float4*)(feat + (size_t)n * HH + lane * 4) = acc;
}

// ---------------- output head + log_softmax ----------------
__global__ __launch_bounds__(256) void head_k(const float* __restrict__ feat,
                                              const float* __restrict__ w,
                                              const float* __restrict__ b,
                                              float* __restrict__ out,
                                              long off0, long off1, long off2,
                                              long probOff) {
    int n = blockIdx.x * 8 + (threadIdx.x >> 5);
    if (n >= NN) return;
    int lane = threadIdx.x & 31;
    float4 f = *(const float4*)(feat + (size_t)n * HH + lane * 4);

    float o[CC];
    float mx = -1e30f;
#pragma unroll
    for (int c = 0; c < CC; ++c) {
        float4 wc = *(const float4*)(w + c * HH + lane * 4);
        float p = f.x * wc.x + f.y * wc.y + f.z * wc.z + f.w * wc.w;
        for (int s = 16; s > 0; s >>= 1) p += __shfl_xor(p, s, 32);
        p += b[c];
        o[c] = p;
        mx = p > mx ? p : mx;
    }
    float den = 0.0f;
#pragma unroll
    for (int c = 0; c < CC; ++c) den += __expf(o[c] - mx);
    float lsm = mx + __logf(den);

    if (lane < CC) {
        float oc = o[lane];
        long base = (long)n * CC + lane;
        out[off0 + base] = oc;
        if (off1 >= 0) out[off1 + base] = oc;
        if (off2 >= 0) out[off2 + base] = oc;
        out[probOff + base] = oc - lsm;
    }
}

// ---------------- host ----------------
static inline size_t align256(size_t x) { return (x + 255) & ~(size_t)255; }

extern "C" void kernel_launch(void* const* d_in, const int* in_sizes, int n_in,
                              void* d_out, int out_size, void* d_ws, size_t ws_size,
                              hipStream_t stream) {
    (void)in_sizes; (void)n_in; (void)out_size; (void)ws_size;
    const float* x   = (const float*)d_in[0];
    const int* ei    = (const int*)d_in[1];   // [2,E]
    const int* et    = (const int*)d_in[2];
    const float* lW  = (const float*)d_in[3];   // [2,R,H,H]
    const float* lRt = (const float*)d_in[4];   // [2,H,H]
    const float* lB  = (const float*)d_in[5];   // [2,H]
    const float* gW  = (const float*)d_in[6];   // [3,R,H,H]
    const float* gRt = (const float*)d_in[7];   // [3,H,H]
    const float* gB  = (const float*)d_in[8];   // [3,H]
    const float* law = (const float*)d_in[9];
    const float* lab = (const float*)d_in[10];
    const float* gaw = (const float*)d_in[11];
    const float* gab = (const float*)d_in[12];
    const float* low = (const float*)d_in[13];
    const float* lob = (const float*)d_in[14];
    const float* gow = (const float*)d_in[15];
    const float* gob = (const float*)d_in[16];
    float* out = (float*)d_out;

    const int* src = ei;
    const int* dst = ei + EE;

    // workspace carve-up
    char* ws = (char*)d_ws;
    size_t off = 0;
    float* invc = (float*)(ws + off); off += align256((size_t)NN * RREL * 4);
    float* msg  = (float*)(ws + off); off += align256((size_t)NN * RREL * HH * 4);
    unsigned short* Bfrag = (unsigned short*)(ws + off); off += align256((size_t)KTOT * HH * 2);
    float* hL1 = (float*)(ws + off); off += align256((size_t)NN * HH * 4);
    float* hL2 = (float*)(ws + off); off += align256((size_t)NN * HH * 4);
    float* hG1 = (float*)(ws + off); off += align256((size_t)NN * HH * 4);
    float* hG2 = (float*)(ws + off); off += align256((size_t)NN * HH * 4);
    float* hG3 = (float*)(ws + off); off += align256((size_t)NN * HH * 4);
    float* featL = msg;              // msg region recycled after layers
    float* featG = msg + (size_t)NN * HH;

    const int TB = 256;
    const int gCnt  = (EE + TB - 1) / TB;
    const int gInv  = (NN * RREL + TB - 1) / TB;
    const int gScat = (EE + 7) / 8;
    const int gWf   = (KTOT * HH + TB - 1) / TB;
    const int gGemm = NN / 16;           // 3125, exact
    const int gNode = (NN + 7) / 8;      // 6250, exact
    const size_t msgBytes = (size_t)NN * RREL * HH * 4;

    // 1) per-(node,relation) counts -> inv_cnt (in place)
    hipMemsetAsync(invc, 0, (size_t)NN * RREL * 4, stream);
    count_k<<<gCnt, TB, 0, stream>>>(dst, et, invc);
    inv_k<<<gInv, TB, 0, stream>>>(invc);

    // 2) msg(x) shared by local layer 0 and global layer 0
    hipMemsetAsync(msg, 0, msgBytes, stream);
    scatter_k<<<gScat, TB, 0, stream>>>(x, src, dst, et, msg);

    wfrag_k<<<gWf, TB, 0, stream>>>(lRt, lW, Bfrag);
    rgcn_gemm_k<<<gGemm, TB, 0, stream>>>(x, msg, invc, Bfrag, lB, hL1);
    wfrag_k<<<gWf, TB, 0, stream>>>(gRt, gW, Bfrag);
    rgcn_gemm_k<<<gGemm, TB, 0, stream>>>(x, msg, invc, Bfrag, gB, hG1);

    // 3) local layer 1
    hipMemsetAsync(msg, 0, msgBytes, stream);
    scatter_k<<<gScat, TB, 0, stream>>>(hL1, src, dst, et, msg);
    wfrag_k<<<gWf, TB, 0, stream>>>(lRt + (size_t)HH * HH, lW + (size_t)RREL * HH * HH, Bfrag);
    rgcn_gemm_k<<<gGemm, TB, 0, stream>>>(hL1, msg, invc, Bfrag, lB + HH, hL2);

    // 4) global layer 1
    hipMemsetAsync(msg, 0, msgBytes, stream);
    scatter_k<<<gScat, TB, 0, stream>>>(hG1, src, dst, et, msg);
    wfrag_k<<<gWf, TB, 0, stream>>>(gRt + (size_t)HH * HH, gW + (size_t)RREL * HH * HH, Bfrag);
    rgcn_gemm_k<<<gGemm, TB, 0, stream>>>(hG1, msg, invc, Bfrag, gB + HH, hG2);

    // 5) global layer 2
    hipMemsetAsync(msg, 0, msgBytes, stream);
    scatter_k<<<gScat, TB, 0, stream>>>(hG2, src, dst, et, msg);
    wfrag_k<<<gWf, TB, 0, stream>>>(gRt + (size_t)2 * HH * HH, gW + (size_t)2 * RREL * HH * HH, Bfrag);
    rgcn_gemm_k<<<gGemm, TB, 0, stream>>>(hG2, msg, invc, Bfrag, gB + 2 * HH, hG3);

    // 6) layer attention (msg region reused as feature buffers)
    attn_k<<<gNode, TB, 0, stream>>>(hL1, hL2, hL2, law, lab, featL, 2);
    attn_k<<<gNode, TB, 0, stream>>>(hG1, hG2, hG3, gaw, gab, featG, 3);

    // 7) heads: outputs = (local_out, global_out, prob_local, global_out, global_out, prob_global)
    head_k<<<gNode, TB, 0, stream>>>(featL, low, lob, out,
                                     (long)0, (long)-1, (long)-1, (long)(2L * NC));
    head_k<<<gNode, TB, 0, stream>>>(featG, gow, gob, out,
                                     (long)(1L * NC), (long)(3L * NC), (long)(4L * NC),
                                     (long)(5L * NC));
}
